// IDAdapterPostfuseModule_16243566313663
// MI455X (gfx1250) — matrix operand order repaired and checked by reading, not compiled, not run on hardware
//
#include <hip/hip_runtime.h>
#include <hip/hip_bf16.h>

#define BB 512
#define SS 77
#define DD 2048
#define MAXOBJN 4
#define EPSF 1e-5f

typedef __attribute__((ext_vector_type(16))) __bf16 v16bf;
typedef __attribute__((ext_vector_type(8)))  float  v8f;

__device__ __forceinline__ unsigned short f2bf(float f) {
    union { float f; unsigned int u; } c; c.f = f;
    unsigned int u = c.u;
    unsigned int r = u + 0x7FFFu + ((u >> 16) & 1u);   // round-to-nearest-even
    return (unsigned short)(r >> 16);
}

// ---------------------------------------------------------------------------
// Build x = concat(tok, mean_obj): one block per batch row.
// ---------------------------------------------------------------------------
__global__ void __launch_bounds__(256)
build_x_kernel(const float* __restrict__ embeds,       // B,S,D
               const float* __restrict__ obj,          // B,4,1,D
               const unsigned char* __restrict__ mask, // B,S (bool)
               const int* __restrict__ nobj,           // B
               float* __restrict__ x,                  // B,2D
               float* __restrict__ tok,                // B,D
               int* __restrict__ idxbuf) {             // B
    int b = blockIdx.x;
    __shared__ int sidx;
    if (threadIdx.x == 0) sidx = 0;
    __syncthreads();
    if (threadIdx.x < SS && mask[(size_t)b * SS + threadIdx.x])
        atomicExch(&sidx, (int)threadIdx.x);           // exactly one true per row
    __syncthreads();
    int idx = sidx;
    if (threadIdx.x == 0) idxbuf[b] = idx;
    int n = nobj[b];
    float inv = 1.0f / (float)n;
    const float* trow = embeds + ((size_t)b * SS + idx) * DD;
    for (int d = threadIdx.x; d < DD; d += blockDim.x) {
        float t = trow[d];
        float s = 0.f;
        for (int o = 0; o < MAXOBJN; ++o)
            if (o < n) s += obj[((size_t)b * MAXOBJN + o) * DD + d];
        x[(size_t)b * (2 * DD) + d]      = t;
        x[(size_t)b * (2 * DD) + DD + d] = s * inv;
        tok[(size_t)b * DD + d]          = t;
    }
}

// ---------------------------------------------------------------------------
// Row LayerNorm -> bf16 (one block per row)
// ---------------------------------------------------------------------------
template <int COLS>
__global__ void __launch_bounds__(256)
ln_bf16_kernel(const float* __restrict__ in,
               const float* __restrict__ g,
               const float* __restrict__ be,
               unsigned short* __restrict__ out) {
    int row = blockIdx.x;
    const float* p = in + (size_t)row * COLS;
    __shared__ float s1[256], s2[256];
    float sum = 0.f, sq = 0.f;
    for (int c = threadIdx.x; c < COLS; c += 256) { float v = p[c]; sum += v; sq += v * v; }
    s1[threadIdx.x] = sum; s2[threadIdx.x] = sq;
    __syncthreads();
    for (int off = 128; off > 0; off >>= 1) {
        if (threadIdx.x < off) {
            s1[threadIdx.x] += s1[threadIdx.x + off];
            s2[threadIdx.x] += s2[threadIdx.x + off];
        }
        __syncthreads();
    }
    float mu  = s1[0] / (float)COLS;
    float var = s2[0] / (float)COLS - mu * mu;
    float inv = rsqrtf(var + EPSF);
    for (int c = threadIdx.x; c < COLS; c += 256) {
        float v = (p[c] - mu) * inv * g[c] + be[c];
        out[(size_t)row * COLS + c] = f2bf(v);
    }
}

// ---------------------------------------------------------------------------
// Pack fp32 weight (K x N row-major) into WMMA-B bf16 layout:
// [K/32][N/16][lane 0..31][16 elems]; lane L: n = nt*16+(L&15),
// K = kt*32 + 16*(L>=16) + e  (contiguous k order, same as the GEMM's A loads).
// ---------------------------------------------------------------------------
__global__ void __launch_bounds__(256)
pack_w_bf16_kernel(const float* __restrict__ W, unsigned short* __restrict__ out,
                   int K, int N) {
    int total = (K / 32) * (N / 16) * 32;
    int ntiles = N / 16;
    for (int i = blockIdx.x * blockDim.x + threadIdx.x; i < total;
         i += gridDim.x * blockDim.x) {
        int lane = i & 31;
        int t    = i >> 5;
        int nt   = t % ntiles;
        int kt   = t / ntiles;
        int n    = nt * 16 + (lane & 15);
        int kb   = kt * 32 + ((lane >> 4) * 16);
        unsigned short* o = out + (size_t)i * 16;
#pragma unroll
        for (int e = 0; e < 16; ++e) o[e] = f2bf(W[(size_t)(kb + e) * N + n]);
    }
}

// ---------------------------------------------------------------------------
// WMMA GEMM: C(MxN,f32) = A(MxK bf16 row-major) x Bpacked. One wave computes a
// 16x64 strip (4 n-tiles reuse the A fragment). 8 waves / block.
// ---------------------------------------------------------------------------
union U16 { v16bf v; uint4 q[2]; };

__global__ void __launch_bounds__(256)
wmma_gemm_kernel(const unsigned short* __restrict__ A,
                 const unsigned short* __restrict__ Bp,
                 float* __restrict__ C,
                 int M, int K, int N) {
    const int lane    = threadIdx.x & 31;
    const int wave    = threadIdx.x >> 5;
    const int ngroups = N >> 6;                      // groups of 4 n-tiles
    const int group   = blockIdx.x * 8 + wave;
    const int mt      = group / ngroups;
    const int ng      = group % ngroups;
    if (mt >= (M >> 4)) return;                      // wave-uniform guard
    const int nt0     = ng * 4;
    const int ntilesN = N >> 4;

    v8f acc0 = {}, acc1 = {}, acc2 = {}, acc3 = {};
    const int m    = mt * 16 + (lane & 15);
    const int kSub = (lane >> 4) * 16;               // contiguous k order (matches packed B)
    const unsigned short* arow = A + (size_t)m * K;

    for (int kt = 0; kt < (K >> 5); ++kt) {
        U16 a;
        const unsigned short* ap = arow + kt * 32 + kSub;
        a.q[0] = *(const uint4*)(ap);                // K = kSub .. kSub+7
        a.q[1] = *(const uint4*)(ap + 8);            // K = kSub+8 .. kSub+15

        const unsigned short* bp =
            Bp + ((size_t)kt * ntilesN + nt0) * 512 + (size_t)lane * 16;
        U16 b0, b1, b2, b3;
        b0.q[0] = *(const uint4*)(bp);          b0.q[1] = *(const uint4*)(bp + 8);
        b1.q[0] = *(const uint4*)(bp + 512);    b1.q[1] = *(const uint4*)(bp + 520);
        b2.q[0] = *(const uint4*)(bp + 1024);   b2.q[1] = *(const uint4*)(bp + 1032);
        b3.q[0] = *(const uint4*)(bp + 1536);   b3.q[1] = *(const uint4*)(bp + 1544);

        acc0 = __builtin_amdgcn_wmma_f32_16x16x32_bf16(false, a.v, false, b0.v,
                                                       (short)0, acc0, false, false);
        acc1 = __builtin_amdgcn_wmma_f32_16x16x32_bf16(false, a.v, false, b1.v,
                                                       (short)0, acc1, false, false);
        acc2 = __builtin_amdgcn_wmma_f32_16x16x32_bf16(false, a.v, false, b2.v,
                                                       (short)0, acc2, false, false);
        acc3 = __builtin_amdgcn_wmma_f32_16x16x32_bf16(false, a.v, false, b3.v,
                                                       (short)0, acc3, false, false);
    }

    // C/D layout: VGPR r -> M = mt*16 + r + 8*(lane>=16), N = nt*16 + (lane&15)
    const int row = mt * 16 + ((lane >> 4) * 8);
    const int col = lane & 15;
#pragma unroll
    for (int r = 0; r < 8; ++r) {
        float* crow = C + (size_t)(row + r) * N + (size_t)nt0 * 16 + col;
        crow[0]  = acc0[r];
        crow[16] = acc1[r];
        crow[32] = acc2[r];
        crow[48] = acc3[r];
    }
}

// ---------------------------------------------------------------------------
// Elementwise epilogues
// ---------------------------------------------------------------------------
__global__ void __launch_bounds__(256)
bias_gelu_bf16_kernel(const float* __restrict__ C, const float* __restrict__ bias,
                      unsigned short* __restrict__ out, int total, int N) {
    for (int i = blockIdx.x * blockDim.x + threadIdx.x; i < total;
         i += gridDim.x * blockDim.x) {
        float v = C[i] + bias[i & (N - 1)];
        float g = 0.5f * v * (1.0f + erff(v * 0.70710678118654752f));  // exact GELU
        out[i] = f2bf(g);
    }
}

__global__ void __launch_bounds__(256)
add_bias_res_kernel(const float* __restrict__ C, const float* __restrict__ bias,
                    const float* __restrict__ res, float* __restrict__ out,
                    int total, int N) {
    for (int i = blockIdx.x * blockDim.x + threadIdx.x; i < total;
         i += gridDim.x * blockDim.x)
        out[i] = C[i] + bias[i & (N - 1)] + res[i];
}

// ---------------------------------------------------------------------------
// Final: t = C4 + b2 + h_resid ; LN(t) ; scatter into out[b, idx, :]
// ---------------------------------------------------------------------------
__global__ void __launch_bounds__(256)
final_ln_scatter_kernel(const float* __restrict__ C4, const float* __restrict__ b2,
                        const float* __restrict__ h, const float* __restrict__ g,
                        const float* __restrict__ be, const int* __restrict__ idxbuf,
                        float* __restrict__ out) {
    int b = blockIdx.x;
    __shared__ float s1[256], s2[256];
    float sum = 0.f, sq = 0.f;
    for (int d = threadIdx.x; d < DD; d += 256) {
        float v = C4[(size_t)b * DD + d] + b2[d] + h[(size_t)b * DD + d];
        sum += v; sq += v * v;
    }
    s1[threadIdx.x] = sum; s2[threadIdx.x] = sq;
    __syncthreads();
    for (int off = 128; off > 0; off >>= 1) {
        if (threadIdx.x < off) {
            s1[threadIdx.x] += s1[threadIdx.x + off];
            s2[threadIdx.x] += s2[threadIdx.x + off];
        }
        __syncthreads();
    }
    float mu  = s1[0] / (float)DD;
    float var = s2[0] / (float)DD - mu * mu;
    float inv = rsqrtf(var + EPSF);
    int idx = idxbuf[b];
    float* orow = out + ((size_t)b * SS + idx) * DD;
    for (int d = threadIdx.x; d < DD; d += 256) {
        float v = C4[(size_t)b * DD + d] + b2[d] + h[(size_t)b * DD + d];
        orow[d] = (v - mu) * inv * g[d] + be[d];
    }
}

// ---------------------------------------------------------------------------
extern "C" void kernel_launch(void* const* d_in, const int* in_sizes, int n_in,
                              void* d_out, int out_size, void* d_ws, size_t ws_size,
                              hipStream_t stream) {
    const float* embeds = (const float*)d_in[0];
    const float* obj    = (const float*)d_in[1];
    const float* ln1g   = (const float*)d_in[2];
    const float* ln1b   = (const float*)d_in[3];
    const float* w1a    = (const float*)d_in[4];
    const float* b1a    = (const float*)d_in[5];
    const float* w2a    = (const float*)d_in[6];
    const float* b2a    = (const float*)d_in[7];
    const float* ln2g   = (const float*)d_in[8];
    const float* ln2b   = (const float*)d_in[9];
    const float* w1b    = (const float*)d_in[10];
    const float* b1b    = (const float*)d_in[11];
    const float* w2b    = (const float*)d_in[12];
    const float* b2b    = (const float*)d_in[13];
    const float* lng    = (const float*)d_in[14];
    const float* lnb    = (const float*)d_in[15];
    const unsigned char* mask = (const unsigned char*)d_in[16];  // bool mask
    const int* nobj     = (const int*)d_in[17];
    float* out = (float*)d_out;

    // Carve workspace
    char* w = (char*)d_ws;
    size_t off = 0;
    auto carve = [&](size_t bytes) -> void* {
        void* p = w + off;
        off += (bytes + 255) & ~(size_t)255;
        return p;
    };
    unsigned short* w1ap = (unsigned short*)carve((size_t)4096 * 2048 * 2);
    unsigned short* w2ap = (unsigned short*)carve((size_t)2048 * 2048 * 2);
    unsigned short* w1bp = (unsigned short*)carve((size_t)2048 * 2048 * 2);
    unsigned short* w2bp = (unsigned short*)carve((size_t)2048 * 2048 * 2);
    float*          xbuf = (float*)carve((size_t)BB * 4096 * 4);
    float*          tok  = (float*)carve((size_t)BB * DD * 4);
    unsigned short* xn   = (unsigned short*)carve((size_t)BB * 4096 * 2);
    float*          Cbuf = (float*)carve((size_t)BB * DD * 4);
    unsigned short* gbuf = (unsigned short*)carve((size_t)BB * DD * 2);
    float*          hbuf = (float*)carve((size_t)BB * DD * 4);
    unsigned short* hn   = (unsigned short*)carve((size_t)BB * DD * 2);
    int*            idxb = (int*)carve((size_t)BB * 4);

    // 1) Bulk copy input -> output (dominant HBM traffic, ~646 MB round trip)
    hipMemcpyAsync(out, embeds, (size_t)BB * SS * DD * sizeof(float),
                   hipMemcpyDeviceToDevice, stream);

    // 2) Pack weights fp32 -> bf16 WMMA-B layout (L2-resident afterwards)
    pack_w_bf16_kernel<<<2048, 256, 0, stream>>>(w1a, w1ap, 4096, 2048);
    pack_w_bf16_kernel<<<1024, 256, 0, stream>>>(w2a, w2ap, 2048, 2048);
    pack_w_bf16_kernel<<<1024, 256, 0, stream>>>(w1b, w1bp, 2048, 2048);
    pack_w_bf16_kernel<<<1024, 256, 0, stream>>>(w2b, w2bp, 2048, 2048);

    // 3) Gather token + mean objects + concat
    build_x_kernel<<<BB, 256, 0, stream>>>(embeds, obj, mask, nobj, xbuf, tok, idxb);

    // 4) MLP1: LN -> GEMM(512x4096x2048) -> GELU -> GEMM(512x2048x2048) -> +tok
    ln_bf16_kernel<4096><<<BB, 256, 0, stream>>>(xbuf, ln1g, ln1b, xn);
    wmma_gemm_kernel<<<128, 256, 0, stream>>>(xn, w1ap, Cbuf, BB, 4096, 2048);
    bias_gelu_bf16_kernel<<<1024, 256, 0, stream>>>(Cbuf, b1a, gbuf, BB * DD, DD);
    wmma_gemm_kernel<<<128, 256, 0, stream>>>(gbuf, w2ap, Cbuf, BB, 2048, 2048);
    add_bias_res_kernel<<<1024, 256, 0, stream>>>(Cbuf, b2a, tok, hbuf, BB * DD, DD);

    // 5) MLP2 with residual
    ln_bf16_kernel<2048><<<BB, 256, 0, stream>>>(hbuf, ln2g, ln2b, hn);
    wmma_gemm_kernel<<<128, 256, 0, stream>>>(hn, w1bp, Cbuf, BB, 2048, 2048);
    bias_gelu_bf16_kernel<<<1024, 256, 0, stream>>>(Cbuf, b1b, gbuf, BB * DD, DD);
    wmma_gemm_kernel<<<128, 256, 0, stream>>>(gbuf, w2bp, Cbuf, BB, 2048, 2048);

    // 6) Final add + LN + scatter into the copied output
    final_ln_scatter_kernel<<<BB, 256, 0, stream>>>(Cbuf, b2b, hbuf, lng, lnb,
                                                    idxb, out);
}
